// FourDSSM_60550448939374
// MI455X (gfx1250) — matrix operand order repaired and checked by reading, not compiled, not run on hardware
//
#include <hip/hip_runtime.h>
#include <hip/hip_fp16.h>

typedef __attribute__((ext_vector_type(16))) _Float16     v16h;
typedef __attribute__((ext_vector_type(8)))  float        v8f;
typedef __attribute__((ext_vector_type(2)))  float        v2f;
typedef __attribute__((ext_vector_type(8)))  unsigned int v8u;

#define BB 4096
#define TTS 512
#define II 4
#define HH 32
#define SS 4

__device__ __forceinline__ float fast_sigmoid(float x) {
    return 1.0f / (1.0f + __expf(-x));
}
__device__ __forceinline__ float fast_tanh(float x) {
    // tanh(x) = 1 - 2/(exp(2x)+1)
    return 1.0f - 2.0f / (__expf(2.0f * x) + 1.0f);
}

__device__ __forceinline__ v8f wmma_f32_k4(v2f a, v2f b, v8f c) {
    return __builtin_amdgcn_wmma_f32_16x16x4_f32(false, a, false, b, (short)0, c, false, false);
}
__device__ __forceinline__ v8f wmma_f16_k32(v16h a, v16h b, v8f c) {
    return __builtin_amdgcn_wmma_f32_16x16x32_f16(false, a, false, b, (short)0, c, false, false);
}

__device__ __forceinline__ unsigned pkf16(float a, float b) {
    auto ph = __builtin_amdgcn_cvt_pkrtz(a, b);   // half2, lowers to v_cvt_pk_rtz_f16_f32
    return __builtin_bit_cast(unsigned, ph);
}

// Build f16 B-operand (32xK x 16N) for 16x16x32 WMMA from two f32 C/D-layout
// tiles holding h^T rows 0-15 (d0) and 16-31 (d1).
//   B vgpr j   : lanes0-15 = pack(h[2j],h[2j+1])     lanes16-31 = pack(h[8+2j],h[9+2j])
//   B vgpr 4+j : lanes0-15 = pack(h[16+2j],h[17+2j]) lanes16-31 = pack(h[24+2j],h[25+2j])
__device__ __forceinline__ v16h build_hB(const v8f& d0, const v8f& d1, bool hi) {
    v8u u;
#pragma unroll
    for (int j = 0; j < 4; ++j) {
        unsigned p0 = pkf16(d0[2 * j], d0[2 * j + 1]); // lo: rows 2j,2j+1   hi: rows 8+2j,9+2j
        unsigned p1 = pkf16(d1[2 * j], d1[2 * j + 1]); // lo: rows 16+..     hi: rows 24+..
        unsigned s0 = (unsigned)__shfl_xor((int)p0, 16);
        unsigned s1 = (unsigned)__shfl_xor((int)p1, 16);
        u[j]     = p0;
        u[4 + j] = p1;
    }
    return __builtin_bit_cast(v16h, u);
}

__global__ __launch_bounds__(256, 1)
void gru_proj_fused(const float* __restrict__ x, const float* __restrict__ h0,
                    const float* __restrict__ w_ih, const float* __restrict__ w_hh,
                    const float* __restrict__ b_ih, const float* __restrict__ b_hh,
                    const float* __restrict__ w_proj, const float* __restrict__ b_proj,
                    float* __restrict__ out_state, float* __restrict__ out_hn,
                    float* __restrict__ dump) {
    const int lane  = (int)(threadIdx.x & 31u);
    const int n     = lane & 15;
    const bool hi   = lane >= 16;
    const int wave  = (int)((blockIdx.x * blockDim.x + threadIdx.x) >> 5);
    const int b0    = wave * 16;
    const int khalf = hi ? 8 : 0;   // K offset for A/B f16 layouts
    const int rsh   = hi ? 8 : 0;   // row offset for C/D layouts

    // ---- Preload W_hh as 6 f16 A-tiles (16 rows x K32 each) -------------
    v8u a_hh_u[6];
#pragma unroll
    for (int tt = 0; tt < 6; ++tt) {
#pragma unroll
        for (int j = 0; j < 8; ++j) {
            int kk = ((j & 3) * 2) + khalf + ((j >> 2) * 16);
            const float* p = w_hh + (size_t)(tt * 16 + n) * HH + kk;
            a_hh_u[tt][j] = pkf16(p[0], p[1]);
        }
    }
    v16h W0 = __builtin_bit_cast(v16h, a_hh_u[0]);
    v16h W1 = __builtin_bit_cast(v16h, a_hh_u[1]);
    v16h W2 = __builtin_bit_cast(v16h, a_hh_u[2]);
    v16h W3 = __builtin_bit_cast(v16h, a_hh_u[3]);
    v16h W4 = __builtin_bit_cast(v16h, a_hh_u[4]);
    v16h W5 = __builtin_bit_cast(v16h, a_hh_u[5]);

    // ---- W_proj as one f16 A-tile, rows 4..15 zero-padded ---------------
    v8u a_pj_u;
#pragma unroll
    for (int j = 0; j < 8; ++j) {
        int kk = ((j & 3) * 2) + khalf + ((j >> 2) * 16);
        float v0 = (n < 4) ? w_proj[(size_t)n * HH + kk]     : 0.0f;
        float v1 = (n < 4) ? w_proj[(size_t)n * HH + kk + 1] : 0.0f;
        a_pj_u[j] = pkf16(v0, v1);
    }
    v16h WP = __builtin_bit_cast(v16h, a_pj_u);

    // ---- W_ih as 6 f32 A-tiles (16 rows x K4) ---------------------------
    v2f a_ih[6];
#pragma unroll
    for (int tt = 0; tt < 6; ++tt) {
        int kof = hi ? 2 : 0;
        a_ih[tt][0] = w_ih[(size_t)(tt * 16 + n) * II + kof];
        a_ih[tt][1] = w_ih[(size_t)(tt * 16 + n) * II + kof + 1];
    }

    // ---- Bias tiles in C/D layout (value depends only on row M) ---------
    v8f biasR0, biasR1, biasZ0, biasZ1, biasXN0, biasXN1, biasHN0, biasHN1, biasPJ;
#pragma unroll
    for (int v = 0; v < 8; ++v) {
        int g = v + rsh;
        biasR0[v]  = b_ih[g]        + b_hh[g];
        biasR1[v]  = b_ih[16 + g]   + b_hh[16 + g];
        biasZ0[v]  = b_ih[32 + g]   + b_hh[32 + g];
        biasZ1[v]  = b_ih[48 + g]   + b_hh[48 + g];
        biasXN0[v] = b_ih[64 + g];
        biasXN1[v] = b_ih[80 + g];
        biasHN0[v] = b_hh[64 + g];
        biasHN1[v] = b_hh[80 + g];
        biasPJ[v]  = (!hi && v < 4) ? b_proj[v] : 0.0f;
    }

    // ---- h state in C/D layout (h^T rows 0-15 / 16-31) ------------------
    v8f h0a, h1a;
#pragma unroll
    for (int v = 0; v < 8; ++v) {
        h0a[v] = h0[(size_t)(b0 + n) * HH + v + rsh];
        h1a[v] = h0[(size_t)(b0 + n) * HH + 16 + v + rsh];
    }
    v16h hB = build_hB(h0a, h1a, hi);

    // per-lane x pointer: B-layout for K=4 f32 WMMA (lane<16 -> i=0,1 ; lane>=16 -> i=2,3)
    const float* xp = x + (size_t)(b0 + n) * TTS * II + (hi ? 2 : 0);

    // per-step state store: lo lanes -> real output (s0..s3 packed b128), hi lanes -> scratch dump
    float4* stp = hi ? (((float4*)dump) + (wave * 16 + n))
                     : (float4*)(out_state + (size_t)(b0 + n) * TTS * SS);
    const int stp_stride = hi ? 0 : 1;

#pragma unroll 1
    for (int t = 0; t < TTS; ++t) {
        float2 xv = *(const float2*)xp;   // global_load_b64
        xp += II;
        v2f xB; xB[0] = xv.x; xB[1] = xv.y;

        // gates^T [96x16] = W_ih[96x4] x x^T[4x16]  (f32 WMMA, bias folded into C)
        v8f aR0  = wmma_f32_k4(a_ih[0], xB, biasR0);
        v8f aR1  = wmma_f32_k4(a_ih[1], xB, biasR1);
        v8f aZ0  = wmma_f32_k4(a_ih[2], xB, biasZ0);
        v8f aZ1  = wmma_f32_k4(a_ih[3], xB, biasZ1);
        v8f aXN0 = wmma_f32_k4(a_ih[4], xB, biasXN0);
        v8f aXN1 = wmma_f32_k4(a_ih[5], xB, biasXN1);
        //          += W_hh[96x32] x h^T[32x16]      (f16 WMMA, f32 accumulate)
        aR0 = wmma_f16_k32(W0, hB, aR0);
        aR1 = wmma_f16_k32(W1, hB, aR1);
        aZ0 = wmma_f16_k32(W2, hB, aZ0);
        aZ1 = wmma_f16_k32(W3, hB, aZ1);
        v8f aHN0 = wmma_f16_k32(W4, hB, biasHN0);  // n-gate hidden part kept separate
        v8f aHN1 = wmma_f16_k32(W5, hB, biasHN1);

        // elementwise GRU update (layouts of all tiles coincide per element)
#pragma unroll
        for (int v = 0; v < 8; ++v) {
            float r0 = fast_sigmoid(aR0[v]);
            float z0 = fast_sigmoid(aZ0[v]);
            float nn0 = fast_tanh(aXN0[v] + r0 * aHN0[v]);
            h0a[v] = nn0 + z0 * (h0a[v] - nn0);

            float r1 = fast_sigmoid(aR1[v]);
            float z1 = fast_sigmoid(aZ1[v]);
            float nn1 = fast_tanh(aXN1[v] + r1 * aHN1[v]);
            h1a[v] = nn1 + z1 * (h1a[v] - nn1);
        }

        // rebuild f16 B operand of h for next step, reuse it for projection
        hB = build_hB(h0a, h1a, hi);

        // state^T [4x16] = W_proj x h^T  (rows 0..3 valid in lo lanes)
        v8f pr = wmma_f16_k32(WP, hB, biasPJ);
        *stp = make_float4(pr[0], pr[1], pr[2], pr[3]);  // global_store_b128
        stp += stp_stride;
    }

    // ---- final hidden state hn [B,H] ------------------------------------
    float* hnp = out_hn + (size_t)(b0 + n) * HH;
#pragma unroll
    for (int v = 0; v < 8; ++v) {
        hnp[v + rsh]      = h0a[v];
        hnp[16 + v + rsh] = h1a[v];
    }
}

extern "C" void kernel_launch(void* const* d_in, const int* in_sizes, int n_in,
                              void* d_out, int out_size, void* d_ws, size_t ws_size,
                              hipStream_t stream) {
    (void)in_sizes; (void)n_in; (void)out_size; (void)ws_size;
    const float* x      = (const float*)d_in[0];
    const float* h0     = (const float*)d_in[1];
    const float* w_ih   = (const float*)d_in[2];
    const float* w_hh   = (const float*)d_in[3];
    const float* b_ih   = (const float*)d_in[4];
    const float* b_hh   = (const float*)d_in[5];
    const float* w_proj = (const float*)d_in[6];
    const float* b_proj = (const float*)d_in[7];

    float* out_state = (float*)d_out;                              // [B,T,S]
    float* out_hn    = out_state + (size_t)BB * TTS * SS;          // [1,B,H]
    float* dump      = (float*)d_ws;                               // hi-lane store sink (64 KB)

    const int waves   = BB / 16;            // 256 waves, one per 16-batch tile
    const int threads = 256;                // 8 waves per block
    const int blocks  = waves * 32 / threads; // 32 blocks
    gru_proj_fused<<<blocks, threads, 0, stream>>>(x, h0, w_ih, w_hh, b_ih, b_hh,
                                                   w_proj, b_proj, out_state, out_hn, dump);
}
